// kMaXTransformerLayer_61572651156147
// MI455X (gfx1250) — matrix operand order repaired and checked by reading, not compiled, not run on hardware
//
#include <hip/hip_runtime.h>
#include <math.h>

// ---------------------------------------------------------------------------
// kMaX transformer layer for MI455X (gfx1250, wave32, WMMA bf16)
// ---------------------------------------------------------------------------

#define SPATIAL 13824   // 24*24*24
#define NB      2

typedef __attribute__((ext_vector_type(16))) __bf16 v16bf;
typedef __attribute__((ext_vector_type(8)))  float  v8f;
typedef __attribute__((ext_vector_type(4)))  unsigned short us4;

__device__ __forceinline__ float gelu_exact(float x) {
    return 0.5f * x * (1.0f + erff(x * 0.70710678118654752440f));
}

__device__ __forceinline__ unsigned short f2bf(float f) {
    union { float f; unsigned u; } a; a.f = f;
    unsigned u = a.u;
    u += 0x7FFFu + ((u >> 16) & 1u);   // round-to-nearest-even
    return (unsigned short)(u >> 16);
}

__device__ __forceinline__ __bf16 bfbits(unsigned short s) {
    union { unsigned short u; __bf16 b; } x; x.u = s; return x.b;
}

// ---------------------------------------------------------------------------
// Full-channel WMMA bf16 GEMM (O == 256):
//   Y[b,o,s] = sum_c act(X[b,c,s]) * W[o,c],  BN stats accumulated.
// One block covers ALL 256 output channels for a 64-wide S slice, so the
// activation tensor is streamed from HBM exactly once (critical for the
// 226MB pixel_feature read). 8 waves x (2 M-tiles x 4 N-frags) = 8 WMMA/step.
// Staging is software-pipelined: global float4 loads for step k+1 are issued
// right after the LDS tiles for step k are published, so HBM latency overlaps
// with the WMMA phase. LDS tiles are pre-swizzled into the CDNA5 fragment
// layout (fragment fetch = 2 contiguous ds_load_b128; A staging packs 4 bf16
// per 8-byte LDS store).
// ---------------------------------------------------------------------------
template <bool GELU_X>
__global__ __launch_bounds__(256)
void gemm_bigM_kernel(const float* __restrict__ X,   // [NB][K][S]
                      const float* __restrict__ Wm,  // [256][K]
                      float* __restrict__ Y,         // [NB][256][S]
                      float* __restrict__ stats,     // [256][2]
                      int K, int S)
{
    __shared__ __align__(32) __bf16 As[16][32][16]; // [m-tile][lane][elem]
    __shared__ __align__(32) __bf16 Bs[4][32][16];  // [n-frag][lane][elem]
    __shared__ float sSum[256], sSq[256];

    const int tid  = threadIdx.x;
    const int b    = blockIdx.z;
    const int s0   = blockIdx.x * 64;
    sSum[tid] = 0.f; sSq[tid] = 0.f;

    const int wv   = tid >> 5;
    const int lane = tid & 31;
    const v8f vz = {};
    v8f c[2][4];
    #pragma unroll
    for (int i = 0; i < 2; ++i)
        #pragma unroll
        for (int j = 0; j < 4; ++j) c[i][j] = vz;

    const float* Xb = X + (size_t)b * K * S;

    // per-thread source pointers and swizzled LDS targets (loop-invariant)
    const float* pA[8];
    __bf16*      dA[8];
    #pragma unroll
    for (int i = 0; i < 8; ++i) {
        int g = i * 256 + tid;            // float4 group id, 0..2047
        int row = g >> 3, cg = g & 7, col = cg * 4;
        pA[i] = Wm + (size_t)row * K + col;
        int h  = (cg >> 1) & 1;
        int eb = (col & 7) + ((col & 16) ? 8 : 0);
        dA[i] = &As[row >> 4][h * 16 + (row & 15)][eb];
    }
    const float* pB[2];
    __bf16*      dB[2];
    #pragma unroll
    for (int i = 0; i < 2; ++i) {
        int g = i * 256 + tid;            // float4 group id, 0..511
        int row = g >> 4, cg = g & 15, col = cg * 4;
        pB[i] = Xb + (size_t)row * S + s0 + col;
        int frag = cg >> 2;
        int lb   = ((row >> 3) & 1) * 16 + (col & 15);
        int e    = (row & 7) + ((row & 16) ? 8 : 0);
        dB[i] = &Bs[frag][lb][e];         // lanes lb..lb+3 are +0/+16/+32/+48
    }

    // prologue: load K-step 0
    float4 ra[8], rb[2];
    #pragma unroll
    for (int i = 0; i < 8; ++i) ra[i] = *(const float4*)pA[i];
    #pragma unroll
    for (int i = 0; i < 2; ++i) rb[i] = *(const float4*)pB[i];

    for (int k0 = 0; k0 < K; k0 += 32) {
        __syncthreads();   // previous compute phase done reading LDS
        // ---- publish current tiles to LDS (convert f32 -> bf16) ----
        #pragma unroll
        for (int i = 0; i < 8; ++i) {
            us4 pk;
            pk[0] = f2bf(ra[i].x); pk[1] = f2bf(ra[i].y);
            pk[2] = f2bf(ra[i].z); pk[3] = f2bf(ra[i].w);
            *(us4*)dA[i] = pk;
        }
        #pragma unroll
        for (int i = 0; i < 2; ++i) {
            float f0 = rb[i].x, f1 = rb[i].y, f2v = rb[i].z, f3 = rb[i].w;
            if (GELU_X) {
                f0 = gelu_exact(f0); f1 = gelu_exact(f1);
                f2v = gelu_exact(f2v); f3 = gelu_exact(f3);
            }
            dB[i][0]  = bfbits(f2bf(f0));
            dB[i][16] = bfbits(f2bf(f1));
            dB[i][32] = bfbits(f2bf(f2v));
            dB[i][48] = bfbits(f2bf(f3));
        }
        __syncthreads();
        // ---- prefetch next K-step (overlaps with WMMA phase below) ----
        if (k0 + 32 < K) {
            #pragma unroll
            for (int i = 0; i < 8; ++i) { pA[i] += 32; ra[i] = *(const float4*)pA[i]; }
            #pragma unroll
            for (int i = 0; i < 2; ++i) { pB[i] += (size_t)32 * S; rb[i] = *(const float4*)pB[i]; }
        }
        // ---- compute ----
        v16bf bfr[4];
        #pragma unroll
        for (int f = 0; f < 4; ++f) bfr[f] = *(const v16bf*)&Bs[f][lane][0];
        #pragma unroll
        for (int mt = 0; mt < 2; ++mt) {
            v16bf a = *(const v16bf*)&As[2 * wv + mt][lane][0];
            #pragma unroll
            for (int f = 0; f < 4; ++f)
                c[mt][f] = __builtin_amdgcn_wmma_f32_16x16x32_bf16(
                    false, a, false, bfr[f], (short)0, c[mt][f], false, false);
        }
    }
    __syncthreads();

    const int half = lane >> 4;
    #pragma unroll
    for (int mt = 0; mt < 2; ++mt) {
        #pragma unroll
        for (int r = 0; r < 8; ++r) {
            int o = wv * 32 + mt * 16 + r + 8 * half;
            float vs = 0.f, vq = 0.f;
            #pragma unroll
            for (int f = 0; f < 4; ++f) {
                float v = c[mt][f][r];
                Y[((size_t)b * 256 + o) * S + s0 + f * 16 + (lane & 15)] = v;
                vs += v; vq += v * v;
            }
            atomicAdd(&sSum[o], vs);
            atomicAdd(&sSq[o],  vq);
        }
    }
    __syncthreads();
    atomicAdd(&stats[tid * 2 + 0], sSum[tid]);
    atomicAdd(&stats[tid * 2 + 1], sSq[tid]);
}

// ---------------------------------------------------------------------------
// Generic WMMA bf16 GEMM, 64(M) x 64(N) tile (used for small-O GEMMs).
// ---------------------------------------------------------------------------
template <bool GELU_X>
__global__ __launch_bounds__(256)
void gemm_bn_kernel(const float* __restrict__ X,    // [NB][K][S]
                    const float* __restrict__ Wm,   // [O][K]
                    const float* __restrict__ bias, // [O] or nullptr
                    float* __restrict__ Y,          // [NB][O][S]
                    float* __restrict__ stats,      // [O][2] or nullptr
                    int K, int O, int S)
{
    __shared__ __align__(32) __bf16 As[4][32][16]; // [m-tile][lane][elem]
    __shared__ __align__(32) __bf16 Bs[4][32][16]; // [n-frag][lane][elem]
    __shared__ float sSum[64], sSq[64];

    const int tid  = threadIdx.x;
    const int b    = blockIdx.z;
    const int s0   = blockIdx.x * 64;
    const int o0   = blockIdx.y * 64;
    if (tid < 64) { sSum[tid] = 0.f; sSq[tid] = 0.f; }

    const int wv   = tid >> 5;
    const int lane = tid & 31;
    const int mi   = wv >> 1;
    const int ni   = wv & 1;
    v8f c0 = {}; v8f c1 = {};

    const float* Xb = X + (size_t)b * K * S;

    for (int k0 = 0; k0 < K; k0 += 32) {
        __syncthreads();
        // stage A (weights) 64x32
        #pragma unroll
        for (int i = 0; i < 2; ++i) {
            int g = i * 256 + tid;             // float4 group, 0..511
            int row = g >> 3, cg = g & 7, col = cg * 4;
            int gro = o0 + row; if (gro >= O) gro = O - 1;
            float4 v = *(const float4*)(Wm + (size_t)gro * K + k0 + col);
            int h  = (cg >> 1) & 1;
            int eb = (col & 7) + ((col & 16) ? 8 : 0);
            us4 pk;
            pk[0] = f2bf(v.x); pk[1] = f2bf(v.y);
            pk[2] = f2bf(v.z); pk[3] = f2bf(v.w);
            *(us4*)&As[row >> 4][h * 16 + (row & 15)][eb] = pk;
        }
        // stage B (activations) 32x64
        #pragma unroll
        for (int i = 0; i < 2; ++i) {
            int g = i * 256 + tid;             // float4 group, 0..511
            int row = g >> 4, cg = g & 15, col = cg * 4;
            float4 v = *(const float4*)(Xb + (size_t)(k0 + row) * S + s0 + col);
            float f0 = v.x, f1 = v.y, f2v = v.z, f3 = v.w;
            if (GELU_X) {
                f0 = gelu_exact(f0); f1 = gelu_exact(f1);
                f2v = gelu_exact(f2v); f3 = gelu_exact(f3);
            }
            int frag = cg >> 2;
            int lb   = ((row >> 3) & 1) * 16 + (col & 15);
            int e    = (row & 7) + ((row & 16) ? 8 : 0);
            Bs[frag][lb + 0][e] = bfbits(f2bf(f0));
            Bs[frag][lb + 1][e] = bfbits(f2bf(f1));
            Bs[frag][lb + 2][e] = bfbits(f2bf(f2v));
            Bs[frag][lb + 3][e] = bfbits(f2bf(f3));
        }
        __syncthreads();
        v16bf a  = *(const v16bf*)&As[mi][lane][0];
        v16bf b0 = *(const v16bf*)&Bs[2 * ni][lane][0];
        v16bf b1 = *(const v16bf*)&Bs[2 * ni + 1][lane][0];
        c0 = __builtin_amdgcn_wmma_f32_16x16x32_bf16(false, a, false, b0,
                                                     (short)0, c0, false, false);
        c1 = __builtin_amdgcn_wmma_f32_16x16x32_bf16(false, a, false, b1,
                                                     (short)0, c1, false, false);
    }
    __syncthreads();

    const int half = lane >> 4;
    const int n0   = s0 + ni * 32 + (lane & 15);
    #pragma unroll
    for (int r = 0; r < 8; ++r) {
        int m = mi * 16 + r + 8 * half;
        int o = o0 + m;
        if (o < O) {
            float v0 = c0[r], v1 = c1[r];
            if (bias) { float bb = bias[o]; v0 += bb; v1 += bb; }
            Y[((size_t)b * O + o) * S + n0]      = v0;
            Y[((size_t)b * O + o) * S + n0 + 16] = v1;
            if (stats) {
                atomicAdd(&sSum[m], v0 + v1);
                atomicAdd(&sSq[m],  v0 * v0 + v1 * v1);
            }
        }
    }
    __syncthreads();
    if (stats && tid < 64 && (o0 + tid) < O) {
        atomicAdd(&stats[(o0 + tid) * 2 + 0], sSum[tid]);
        atomicAdd(&stats[(o0 + tid) * 2 + 1], sSq[tid]);
    }
}

// ---------------------------------------------------------------------------
// BN helpers
// ---------------------------------------------------------------------------
__global__ void bn_finalize_kernel(const float* __restrict__ stats,
                                   const float* __restrict__ g,
                                   const float* __restrict__ bparam,
                                   float* __restrict__ scaleShift,
                                   int O, float count)
{
    int o = blockIdx.x * blockDim.x + threadIdx.x;
    if (o >= O) return;
    float mean = stats[o * 2] / count;
    float var  = stats[o * 2 + 1] / count - mean * mean;
    float sc   = g[o] * rsqrtf(var + 1e-3f);
    scaleShift[o * 2]     = sc;
    scaleShift[o * 2 + 1] = bparam[o] - mean * sc;
}

template <bool GELU>
__global__ void bn_apply_kernel(float* __restrict__ Y,
                                const float* __restrict__ ss,
                                int O, int S, size_t total)
{
    for (size_t i = blockIdx.x * (size_t)blockDim.x + threadIdx.x; i < total;
         i += (size_t)gridDim.x * blockDim.x) {
        int o = (int)((i / (size_t)S) % (size_t)O);
        float v = Y[i] * ss[o * 2] + ss[o * 2 + 1];
        if (GELU) v = gelu_exact(v);
        Y[i] = v;
    }
}

__global__ void zero_kernel(float* __restrict__ p, int n)
{
    int i = blockIdx.x * blockDim.x + threadIdx.x;
    if (i < n) p[i] = 0.f;
}

// ---------------------------------------------------------------------------
// 5x5x5 depthwise conv, pad 2, on (NB,256,24,24,24) + BN stats
// ---------------------------------------------------------------------------
__global__ __launch_bounds__(128)
void dwconv_kernel(const float* __restrict__ X, const float* __restrict__ Wdw,
                   float* __restrict__ Y, float* __restrict__ stats)
{
    __shared__ float wsh[125];
    __shared__ float red[128], red2[128];
    const int c = blockIdx.y, b = blockIdx.z;
    const int t = threadIdx.x;
    const int s = blockIdx.x * 128 + t;
    if (t < 125) wsh[t] = Wdw[c * 125 + t];
    __syncthreads();
    int z = s % 24, y = (s / 24) % 24, x = s / 576;
    const float* Xc = X + ((size_t)b * 256 + c) * SPATIAL;
    float acc = 0.f;
    for (int i = 0; i < 5; ++i) {
        int xx = x + i - 2; if ((unsigned)xx >= 24u) continue;
        for (int j = 0; j < 5; ++j) {
            int yy = y + j - 2; if ((unsigned)yy >= 24u) continue;
            const float* rowp = Xc + xx * 576 + yy * 24;
            const float* wr   = wsh + i * 25 + j * 5;
            #pragma unroll
            for (int k = 0; k < 5; ++k) {
                int zz = z + k - 2; if ((unsigned)zz >= 24u) continue;
                acc += rowp[zz] * wr[k];
            }
        }
    }
    Y[((size_t)b * 256 + c) * SPATIAL + s] = acc;
    red[t] = acc; red2[t] = acc * acc; __syncthreads();
    for (int off = 64; off; off >>= 1) {
        if (t < off) { red[t] += red[t + off]; red2[t] += red2[t + off]; }
        __syncthreads();
    }
    if (t == 0) {
        atomicAdd(&stats[c * 2], red[0]);
        atomicAdd(&stats[c * 2 + 1], red2[0]);
    }
}

// ---------------------------------------------------------------------------
// Tiny 1x1 conv + exact in-block BN over the 64 (b,l) columns.
// Wm==nullptr -> identity (pure BN of X laid out as [NB][O][32]).
// ---------------------------------------------------------------------------
__global__ __launch_bounds__(64)
void tiny_conv_bn_kernel(const float* __restrict__ X, const float* __restrict__ Wm,
                         const float* __restrict__ bias,
                         const float* __restrict__ g, const float* __restrict__ bparam,
                         float* __restrict__ Y, int C, int O, int applyGelu)
{
    __shared__ float r1[64], r2[64];
    __shared__ float mv[2];
    const int o = blockIdx.x, t = threadIdx.x;
    const int b = t >> 5, l = t & 31;
    float acc = bias ? bias[o] : 0.f;
    if (Wm) {
        const float* Xb = X + (size_t)b * C * 32 + l;
        const float* Wo = Wm + (size_t)o * C;
        for (int c = 0; c < C; ++c) acc += Wo[c] * Xb[(size_t)c * 32];
    } else {
        acc += X[((size_t)b * O + o) * 32 + l];
    }
    r1[t] = acc; r2[t] = acc * acc; __syncthreads();
    for (int off = 32; off; off >>= 1) {
        if (t < off) { r1[t] += r1[t + off]; r2[t] += r2[t + off]; }
        __syncthreads();
    }
    if (t == 0) {
        float m = r1[0] * (1.f / 64.f);
        mv[0] = m; mv[1] = r2[0] * (1.f / 64.f) - m * m;
    }
    __syncthreads();
    float out = (acc - mv[0]) * rsqrtf(mv[1] + 1e-3f) * g[o] + bparam[o];
    if (applyGelu) out = gelu_exact(out);
    Y[((size_t)b * O + o) * 32 + l] = out;
}

// ---------------------------------------------------------------------------
// pnorm = x / max(||x||_2 over 16 ch, 1e-12)
// ---------------------------------------------------------------------------
__global__ void pnorm_kernel(const float* __restrict__ X, float* __restrict__ Y)
{
    int i = blockIdx.x * blockDim.x + threadIdx.x;
    if (i >= NB * SPATIAL) return;
    int b = i / SPATIAL, s = i % SPATIAL;
    const float* Xb = X + (size_t)b * 16 * SPATIAL + s;
    float v[16]; float nrm = 0.f;
    #pragma unroll
    for (int c = 0; c < 16; ++c) { v[c] = Xb[(size_t)c * SPATIAL]; nrm += v[c] * v[c]; }
    float d = fmaxf(sqrtf(nrm), 1e-12f);
    float* Yb = Y + (size_t)b * 16 * SPATIAL + s;
    #pragma unroll
    for (int c = 0; c < 16; ++c) Yb[(size_t)c * SPATIAL] = v[c] / d;
}

// ---------------------------------------------------------------------------
// mask_logits[b,n,s] = sum_c pnorm[b,c,s]*mk[b,c,n]  (+ BN stats over (b,s))
// ---------------------------------------------------------------------------
__global__ __launch_bounds__(256)
void mask_logits_kernel(const float* __restrict__ Pn, const float* __restrict__ MK,
                        float* __restrict__ ML, float* __restrict__ stats)
{
    __shared__ float mks[16][32];
    __shared__ float part[32][2];
    const int b = blockIdx.y;
    const int t = threadIdx.x;
    const int s = blockIdx.x * 256 + t;
    if (t < 32) { part[t][0] = 0.f; part[t][1] = 0.f; }
    if (t < 512) { int c = t >> 5, n = t & 31; mks[c][n] = MK[((size_t)b * 16 + c) * 32 + n]; }
    __syncthreads();
    float pv[16];
    const float* Pb = Pn + (size_t)b * 16 * SPATIAL + s;
    #pragma unroll
    for (int c = 0; c < 16; ++c) pv[c] = Pb[(size_t)c * SPATIAL];
    float* Mb = ML + (size_t)b * 32 * SPATIAL + s;
    for (int n = 0; n < 32; ++n) {
        float acc = 0.f;
        #pragma unroll
        for (int c = 0; c < 16; ++c) acc += pv[c] * mks[c][n];
        Mb[(size_t)n * SPATIAL] = acc;
        atomicAdd(&part[n][0], acc);
        atomicAdd(&part[n][1], acc * acc);
    }
    __syncthreads();
    if (t < 32) {
        atomicAdd(&stats[t * 2], part[t][0]);
        atomicAdd(&stats[t * 2 + 1], part[t][1]);
    }
}

// ---------------------------------------------------------------------------
// cls = W_cls @ query_space + bias + void_bias(last class), layout (b,l,o)
// ---------------------------------------------------------------------------
__global__ void cls_kernel(const float* __restrict__ QS, const float* __restrict__ Wc,
                           const float* __restrict__ bc, float* __restrict__ OUT)
{
    int i = blockIdx.x * blockDim.x + threadIdx.x;
    if (i >= NB * 32 * 134) return;
    int o = i % 134, l = (i / 134) % 32, b = i / (134 * 32);
    const float* q = QS + (size_t)b * 256 * 32 + l;
    const float* ww = Wc + (size_t)o * 256;
    float acc = bc[o];
    for (int c = 0; c < 256; ++c) acc += ww[c] * q[(size_t)c * 32];
    if (o == 133) acc += logf(133.0f * 0.9f / 0.1f);
    OUT[i] = acc;
}

__global__ void argmax_kernel(const float* __restrict__ ML, int* __restrict__ IDX)
{
    int i = blockIdx.x * blockDim.x + threadIdx.x;
    if (i >= NB * SPATIAL) return;
    int b = i / SPATIAL, s = i % SPATIAL;
    const float* Mb = ML + (size_t)b * 32 * SPATIAL + s;
    float best = Mb[0]; int bi = 0;
    for (int n = 1; n < 32; ++n) {
        float v = Mb[(size_t)n * SPATIAL];
        if (v > best) { best = v; bi = n; }
    }
    IDX[i] = bi;
}

// ---------------------------------------------------------------------------
// km[b,d,l] += pixel_value[b,d,s] for l = argmax(s).  Thread owns channel d,
// per-voxel index is uniform (LDS broadcast) -> no atomics inside the loop.
// ---------------------------------------------------------------------------
__global__ __launch_bounds__(256)
void km_accum_kernel(const float* __restrict__ PV, const int* __restrict__ IDX,
                     float* __restrict__ KM)
{
    __shared__ float acc[256][33];
    __shared__ int idxs[128];
    const int b = blockIdx.z;
    const int s0 = blockIdx.x * 128;
    const int t = threadIdx.x;
    #pragma unroll
    for (int j = 0; j < 32; ++j) acc[t][j] = 0.f;
    if (t < 128) idxs[t] = IDX[b * SPATIAL + s0 + t];
    __syncthreads();
    const float* Pb = PV + ((size_t)b * 256 + t) * SPATIAL + s0;
    for (int s = 0; s < 128; ++s) acc[t][idxs[s]] += Pb[s];
    __syncthreads();
    #pragma unroll
    for (int j = 0; j < 32; ++j)
        atomicAdd(&KM[((size_t)b * 256 + t) * 32 + j], acc[t][j]);
}

// ---------------------------------------------------------------------------
// attention: logits per (b,h) + per-head BN stats
// ---------------------------------------------------------------------------
__global__ __launch_bounds__(256)
void attn_logits_kernel(const float* __restrict__ QKV, float* __restrict__ LG,
                        float* __restrict__ statsSim)
{
    __shared__ float red[256], red2[256];
    const int bh = blockIdx.x, b = bh >> 3, h = bh & 7;
    const int t = threadIdx.x;
    const float* Q  = QKV + ((size_t)b * 512 + h * 16) * 32;
    const float* Kk = QKV + ((size_t)b * 512 + 128 + h * 16) * 32;
    float lsum = 0.f, lsq = 0.f;
    for (int p = t; p < 1024; p += 256) {
        int l = p >> 5, m = p & 31;
        float acc = 0.f;
        #pragma unroll
        for (int d = 0; d < 16; ++d) acc += Q[d * 32 + l] * Kk[d * 32 + m];
        LG[(size_t)bh * 1024 + p] = acc;
        lsum += acc; lsq += acc * acc;
    }
    red[t] = lsum; red2[t] = lsq; __syncthreads();
    for (int off = 128; off; off >>= 1) {
        if (t < off) { red[t] += red[t + off]; red2[t] += red2[t + off]; }
        __syncthreads();
    }
    if (t == 0) {
        atomicAdd(&statsSim[h * 2], red[0]);
        atomicAdd(&statsSim[h * 2 + 1], red2[0]);
    }
}

__global__ __launch_bounds__(256)
void attn_soft_rv_kernel(const float* __restrict__ LG, const float* __restrict__ ssSim,
                         const float* __restrict__ QKV, float* __restrict__ RV)
{
    __shared__ float at[32][33];
    const int bh = blockIdx.x, b = bh >> 3, h = bh & 7;
    const int t = threadIdx.x;
    const float sc = ssSim[h * 2], sh = ssSim[h * 2 + 1];
    if (t < 32) {
        int l = t;
        float row[32]; float mx = -1e30f;
        for (int m = 0; m < 32; ++m) {
            float v = LG[(size_t)bh * 1024 + l * 32 + m] * sc + sh;
            row[m] = v; mx = fmaxf(mx, v);
        }
        float sum = 0.f;
        for (int m = 0; m < 32; ++m) { float e = expf(row[m] - mx); row[m] = e; sum += e; }
        float inv = 1.f / sum;
        for (int m = 0; m < 32; ++m) at[l][m] = row[m] * inv;
    }
    __syncthreads();
    const float* V = QKV + ((size_t)b * 512 + 256 + h * 32) * 32;
    for (int p = t; p < 1024; p += 256) {
        int d = p >> 5, l = p & 31;
        float acc = 0.f;
        #pragma unroll
        for (int m = 0; m < 32; ++m) acc += at[l][m] * V[d * 32 + m];
        RV[((size_t)b * 256 + h * 32 + d) * 32 + l] = acc;
    }
}

template <bool GELU>
__global__ void add_kernel(const float* __restrict__ A, const float* __restrict__ B,
                           float* __restrict__ Y, int n)
{
    int i = blockIdx.x * blockDim.x + threadIdx.x;
    if (i >= n) return;
    float v = A[i] + B[i];
    if (GELU) v = gelu_exact(v);
    Y[i] = v;
}

// ---------------------------------------------------------------------------
// host side
// ---------------------------------------------------------------------------
extern "C" void kernel_launch(void* const* d_in, const int* in_sizes, int n_in,
                              void* d_out, int out_size, void* d_ws, size_t ws_size,
                              hipStream_t stream)
{
    (void)in_sizes; (void)n_in; (void)out_size; (void)ws_size;
    const float* pf       = (const float*)d_in[0];
    const float* qf       = (const float*)d_in[1];
    const float* w_pixel1 = (const float*)d_in[2];
    const float* g_pixel1 = (const float*)d_in[3];
    const float* b_pixel1 = (const float*)d_in[4];
    const float* w_query1 = (const float*)d_in[5];
    const float* g_query1 = (const float*)d_in[6];
    const float* b_query1 = (const float*)d_in[7];
    const float* w_pixelv = (const float*)d_in[8];
    const float* g_pixelv = (const float*)d_in[9];
    const float* b_pixelv = (const float*)d_in[10];
    const float* w_dw     = (const float*)d_in[11];
    const float* g_dw     = (const float*)d_in[12];
    const float* b_dw     = (const float*)d_in[13];
    const float* w_ph1    = (const float*)d_in[14];
    const float* g_ph1    = (const float*)d_in[15];
    const float* b_ph1    = (const float*)d_in[16];
    const float* w_ph2    = (const float*)d_in[17];
    const float* bias_ph2 = (const float*)d_in[18];
    const float* g_ph2    = (const float*)d_in[19];
    const float* b_ph2    = (const float*)d_in[20];
    const float* w_cls    = (const float*)d_in[21];
    const float* bias_cls = (const float*)d_in[22];
    const float* w_mask   = (const float*)d_in[23];
    const float* g_mask   = (const float*)d_in[24];
    const float* b_mask   = (const float*)d_in[25];
    const float* g_pm     = (const float*)d_in[26];
    const float* b_pm     = (const float*)d_in[27];
    const float* g_kmv    = (const float*)d_in[28];
    const float* b_kmv    = (const float*)d_in[29];
    const float* w_km3    = (const float*)d_in[30];
    const float* g_km3    = (const float*)d_in[31];
    const float* b_km3    = (const float*)d_in[32];
    const float* w_qkv    = (const float*)d_in[33];
    const float* g_qkv    = (const float*)d_in[34];
    const float* b_qkv    = (const float*)d_in[35];
    const float* g_sim    = (const float*)d_in[36];
    const float* b_sim    = (const float*)d_in[37];
    const float* g_rv     = (const float*)d_in[38];
    const float* b_rv     = (const float*)d_in[39];
    const float* w_q3     = (const float*)d_in[40];
    const float* g_q3     = (const float*)d_in[41];
    const float* b_q3     = (const float*)d_in[42];
    const float* w_ff1    = (const float*)d_in[43];
    const float* g_ff1    = (const float*)d_in[44];
    const float* b_ff1    = (const float*)d_in[45];
    const float* w_ff2    = (const float*)d_in[46];
    const float* g_ff2    = (const float*)d_in[47];
    const float* b_ff2    = (const float*)d_in[48];

    const size_t PS = (size_t)256 * SPATIAL * NB;  // 7,077,888
    float* w = (float*)d_ws;
    size_t off = 0;
    auto alloc = [&](size_t n) { float* p = w + off; off += n; return p; };

    float* P1    = alloc(PS);                       // pixel_space
    float* P2    = alloc(PS);                       // pixel_value
    float* P3    = alloc(PS);                       // dw-conv branch
    float* Q1    = alloc((size_t)32 * SPATIAL * NB);
    float* Q2    = alloc((size_t)16 * SPATIAL * NB);
    float* QS    = alloc(16384);                    // query_space
    float* QKV   = alloc(32768);
    float* KMRAW = alloc(16384);
    float* KMBN  = alloc(16384);
    float* KM3   = alloc(16384);
    float* QF1   = alloc(16384);
    float* QF2   = alloc(16384);
    float* F1    = alloc(16384);
    float* F2    = alloc(16384);
    float* LG    = alloc(16384);
    float* RV    = alloc(16384);
    float* RVBN  = alloc(16384);
    float* AUP   = alloc(16384);
    float* MK    = alloc(1024);
    int*   IDX   = (int*)alloc(NB * SPATIAL);
    float* STATS = alloc(8192);                     // 8 slots x 1024
    float* SS    = alloc(8192);

    float* S1 = STATS + 0,    *SS1 = SS + 0;        // pixel1
    float* S2 = STATS + 1024, *SS2 = SS + 1024;     // pixelv
    float* S3 = STATS + 2048, *SS3 = SS + 2048;     // dw
    float* S4 = STATS + 3072, *SS4 = SS + 3072;     // ph1
    float* S5 = STATS + 4096, *SS5 = SS + 4096;     // ph2
    float* S6 = STATS + 5120, *SS6 = SS + 5120;     // pm
    float* SSim = STATS + 6144, *SSsim = SS + 6144; // sim (8 heads)

    float* OUT1 = (float*)d_out;                    // qf       (16384)
    float* OUT2 = OUT1 + 16384;                     // cls      (8576)
    float* OUT3 = OUT2 + 8576;                      // mask_logits (884736)
    float* OUT4 = OUT3 + (size_t)32 * SPATIAL * NB; // pnorm    (442368)

    const float cntSpat = (float)(NB * SPATIAL);    // 27648
    const dim3 gBig(SPATIAL / 64, 1, NB);           // full-channel GEMMs
    const dim3 gGemm1o(SPATIAL / 64, 1, NB);        // small-O GEMMs

    // 0) zero stats + km accumulators
    zero_kernel<<<(8192 + 255) / 256, 256, 0, stream>>>(STATS, 8192);
    zero_kernel<<<(16384 + 255) / 256, 256, 0, stream>>>(KMRAW, 16384);

    // 1) query_space
    tiny_conv_bn_kernel<<<256, 64, 0, stream>>>(qf, w_query1, nullptr,
                                                g_query1, b_query1, QS, 256, 256, 1);
    // 2) pixel_space GEMM (bf16 WMMA, gelu fused on input, pf read ONCE) + BN
    gemm_bigM_kernel<true><<<gBig, 256, 0, stream>>>(pf, w_pixel1, P1, S1,
                                                     2048, SPATIAL);
    bn_finalize_kernel<<<1, 256, 0, stream>>>(S1, g_pixel1, b_pixel1, SS1, 256, cntSpat);
    bn_apply_kernel<true><<<2048, 256, 0, stream>>>(P1, SS1, 256, SPATIAL, PS);

    // 3) depthwise conv branch + BN + gelu
    dwconv_kernel<<<dim3(SPATIAL / 128, 256, NB), 128, 0, stream>>>(P1, w_dw, P3, S3);
    bn_finalize_kernel<<<1, 256, 0, stream>>>(S3, g_dw, b_dw, SS3, 256, cntSpat);
    bn_apply_kernel<true><<<2048, 256, 0, stream>>>(P3, SS3, 256, SPATIAL, PS);

    // 4) ph1, ph2
    gemm_bn_kernel<false><<<gGemm1o, 256, 0, stream>>>(P3, w_ph1, nullptr, Q1, S4,
                                                       256, 32, SPATIAL);
    bn_finalize_kernel<<<1, 256, 0, stream>>>(S4, g_ph1, b_ph1, SS4, 32, cntSpat);
    bn_apply_kernel<true><<<1024, 256, 0, stream>>>(Q1, SS4, 32, SPATIAL,
                                                    (size_t)32 * SPATIAL * NB);
    gemm_bn_kernel<false><<<gGemm1o, 256, 0, stream>>>(Q1, w_ph2, bias_ph2, Q2, S5,
                                                       32, 16, SPATIAL);
    bn_finalize_kernel<<<1, 256, 0, stream>>>(S5, g_ph2, b_ph2, SS5, 16, cntSpat);
    bn_apply_kernel<false><<<1024, 256, 0, stream>>>(Q2, SS5, 16, SPATIAL,
                                                     (size_t)16 * SPATIAL * NB);
    // 5) pnorm -> OUT4
    pnorm_kernel<<<(NB * SPATIAL + 255) / 256, 256, 0, stream>>>(Q2, OUT4);

    // 6) mk, mask logits -> OUT3 (BN in place), cls -> OUT2
    tiny_conv_bn_kernel<<<16, 64, 0, stream>>>(QS, w_mask, nullptr,
                                               g_mask, b_mask, MK, 256, 16, 0);
    mask_logits_kernel<<<dim3(SPATIAL / 256, NB), 256, 0, stream>>>(OUT4, MK, OUT3, S6);
    bn_finalize_kernel<<<1, 256, 0, stream>>>(S6, g_pm, b_pm, SS6, 32, cntSpat);
    bn_apply_kernel<false><<<1024, 256, 0, stream>>>(OUT3, SS6, 32, SPATIAL,
                                                     (size_t)32 * SPATIAL * NB);
    cls_kernel<<<(NB * 32 * 134 + 255) / 256, 256, 0, stream>>>(QS, w_cls, bias_cls, OUT2);

    // 7) hard assignment + pixel_value GEMM + km
    argmax_kernel<<<(NB * SPATIAL + 255) / 256, 256, 0, stream>>>(OUT3, IDX);
    gemm_bigM_kernel<false><<<gBig, 256, 0, stream>>>(P1, w_pixelv, P2, S2,
                                                      256, SPATIAL);
    bn_finalize_kernel<<<1, 256, 0, stream>>>(S2, g_pixelv, b_pixelv, SS2, 256, cntSpat);
    bn_apply_kernel<false><<<2048, 256, 0, stream>>>(P2, SS2, 256, SPATIAL, PS);
    km_accum_kernel<<<dim3(SPATIAL / 128, 1, NB), 256, 0, stream>>>(P2, IDX, KMRAW);
    tiny_conv_bn_kernel<<<256, 64, 0, stream>>>(KMRAW, nullptr, nullptr,
                                                g_kmv, b_kmv, KMBN, 256, 256, 0);
    tiny_conv_bn_kernel<<<256, 64, 0, stream>>>(KMBN, w_km3, nullptr,
                                                g_km3, b_km3, KM3, 256, 256, 0);
    add_kernel<false><<<64, 256, 0, stream>>>(qf, KM3, QF1, 16384);

    // 8) query self-attention
    tiny_conv_bn_kernel<<<512, 64, 0, stream>>>(QS, w_qkv, nullptr,
                                                g_qkv, b_qkv, QKV, 256, 512, 0);
    attn_logits_kernel<<<16, 256, 0, stream>>>(QKV, LG, SSim);
    bn_finalize_kernel<<<1, 256, 0, stream>>>(SSim, g_sim, b_sim, SSsim, 8, 2048.f);
    attn_soft_rv_kernel<<<16, 256, 0, stream>>>(LG, SSsim, QKV, RV);
    tiny_conv_bn_kernel<<<256, 64, 0, stream>>>(RV, nullptr, nullptr,
                                                g_rv, b_rv, RVBN, 256, 256, 1);
    tiny_conv_bn_kernel<<<256, 64, 0, stream>>>(RVBN, w_q3, nullptr,
                                                g_q3, b_q3, AUP, 256, 256, 0);
    add_kernel<true><<<64, 256, 0, stream>>>(QF1, AUP, QF2, 16384);

    // 9) FFN
    tiny_conv_bn_kernel<<<256, 64, 0, stream>>>(QF2, w_ff1, nullptr,
                                                g_ff1, b_ff1, F1, 256, 256, 1);
    tiny_conv_bn_kernel<<<256, 64, 0, stream>>>(F1, w_ff2, nullptr,
                                                g_ff2, b_ff2, F2, 256, 256, 0);
    add_kernel<true><<<64, 256, 0, stream>>>(QF2, F2, OUT1, 16384);
}
